// DeepHedging_NN_42210938585852
// MI455X (gfx1250) — matrix operand order repaired and checked by reading, not compiled, not run on hardware
//
#include <hip/hip_runtime.h>

typedef __attribute__((ext_vector_type(16))) _Float16 v16h;
typedef __attribute__((ext_vector_type(8)))  _Float16 v8h;
typedef __attribute__((ext_vector_type(8)))  float    v8f;
typedef __attribute__((ext_vector_type(4)))  float    v4f;

#define H_DIM   128
#define IN_DIM  16
#define O_DIM   8
#define GIN_DIM (IN_DIM + O_DIM)   // 24
#define NSTEP   64
#define G3      (3 * H_DIM)        // 384
#define NT      (G3 / 16)          // 24 N-tiles of the 384-wide gate output
#define KC      (H_DIM / 32)       // 4 K-chunks of H
#define HROW    136                // padded halves per sim row of h (128 + 8 pad)
#define WAVES   8
#define SPW     16                 // sims per wave (one WMMA M tile)
#define SPB     (WAVES * SPW)      // 128 sims per block

static __device__ inline v8f wmma16(v16h a, v16h b, v8f c) {
  return __builtin_amdgcn_wmma_f32_16x16x32_f16(false, a, false, b, (short)0, c,
                                                false, false);
}

static __device__ inline v16h cat16(v8h lo, v8h hi) {
  return __builtin_shufflevector(lo, hi, 0, 1, 2, 3, 4, 5, 6, 7, 8, 9, 10, 11,
                                 12, 13, 14, 15);
}

// Fragment stored as two 512B planes; each lane does two conflict-free b128 loads.
static __device__ inline v16h load_frag(const _Float16* base, int lane) {
  v8h lo = *(const v8h*)(base + lane * 8);
  v8h hi = *(const v8h*)(base + 256 + lane * 8);
  return cat16(lo, hi);
}

// A-fragment (16x32 f16) of h from padded LDS rows, per ISA 16-bit A layout:
// lane<16 holds M=lane, K = c*32+{0..7, 16..23}; lane>=16 holds K = c*32+{8..15, 24..31}.
static __device__ inline v16h load_hA(const _Float16* hbuf, int lane, int c) {
  const int m = lane & 15;
  const int off = c * 32 + ((lane & 16) ? 8 : 0);
  const _Float16* row = hbuf + m * HROW + off;
  v8h lo = *(const v8h*)(row);
  v8h hi = *(const v8h*)(row + 16);
  return cat16(lo, hi);
}

static __device__ inline float fsig(float x) {
  x = fminf(fmaxf(x, -30.f), 30.f);
  return 1.f / (1.f + __expf(-x));
}
static __device__ inline float ftanh_(float x) {
  x = fminf(fmaxf(x, -15.f), 15.f);
  float e = __expf(-2.f * x);
  return (1.f - e) / (1.f + e);
}

__global__ __launch_bounds__(256, 1) void DeepHedging_GRU_wmma_kernel(
    const float* __restrict__ X, const float* __restrict__ W_ih,
    const float* __restrict__ W_hh, const float* __restrict__ b_ih,
    const float* __restrict__ b_hh, const float* __restrict__ W_out,
    const float* __restrict__ b_out, float* __restrict__ out) {
  __shared__ __align__(16) _Float16 sBhh[NT * KC * 512];   // 96 KB
  __shared__ __align__(16) _Float16 sBih[NT * 512];        // 24 KB
  __shared__ __align__(16) _Float16 sBout[KC * 512];       // 4 KB
  __shared__ __align__(16) _Float16 sH[WAVES * 2 * SPW * HROW];  // 68 KB ping-pong h
  __shared__ __align__(16) _Float16 sPos[WAVES * SPW * O_DIM];   // 2 KB pos transpose
  __shared__ float sBi[G3], sBh[G3], sBo[O_DIM];

  const int tid = threadIdx.x;

  // ---- one-time setup: pre-swizzle weights into WMMA B-fragment layout ----
  // B = W^T (K x N): lane<16 -> N = tile*16+lane, K = chunk*32 + plane*16 + j
  //                  lane>=16 -> same N, K offset +8.
  for (int i = tid; i < NT * KC * 512; i += 256) {
    int frag = i >> 9, r = i & 511;
    int p = r >> 8, l = (r >> 3) & 31, j = r & 7;
    int t = frag >> 2, c = frag & 3;
    int n = t * 16 + (l & 15);
    int k = c * 32 + ((l & 16) ? 8 : 0) + p * 16 + j;
    sBhh[i] = (_Float16)W_hh[n * H_DIM + k];
  }
  for (int i = tid; i < NT * 512; i += 256) {
    int t = i >> 9, r = i & 511;
    int p = r >> 8, l = (r >> 3) & 31, j = r & 7;
    int n = t * 16 + (l & 15);
    int k = ((l & 16) ? 8 : 0) + p * 16 + j;
    sBih[i] = (k < GIN_DIM) ? (_Float16)W_ih[n * GIN_DIM + k] : (_Float16)0.f;
  }
  for (int i = tid; i < KC * 512; i += 256) {
    int c = i >> 9, r = i & 511;
    int p = r >> 8, l = (r >> 3) & 31, j = r & 7;
    int n = l & 15;
    int k = c * 32 + ((l & 16) ? 8 : 0) + p * 16 + j;
    sBout[i] = (n < O_DIM) ? (_Float16)W_out[n * H_DIM + k] : (_Float16)0.f;
  }
  for (int i = tid; i < G3; i += 256) { sBi[i] = b_ih[i]; sBh[i] = b_hh[i]; }
  if (tid < O_DIM) sBo[tid] = b_out[tid];
  for (int i = tid; i < WAVES * 2 * SPW * HROW; i += 256) sH[i] = (_Float16)0.f;
  for (int i = tid; i < WAVES * SPW * O_DIM; i += 256) sPos[i] = (_Float16)0.f;
  __syncthreads();

  const int w = tid >> 5, lane = tid & 31;
  const int m16 = lane & 15;                 // sim row (A) / column (C,D)
  const int mb = (lane & 16) ? 8 : 0;        // D-layout row base for upper lanes
  const int simBase = blockIdx.x * SPB + w * SPW;
  _Float16* hbuf0 = sH + (w * 2 + 0) * SPW * HROW;
  _Float16* hbuf1 = sH + (w * 2 + 1) * SPW * HROW;
  _Float16* posrow = sPos + w * SPW * O_DIM;

  float pos[8];
#pragma unroll
  for (int r = 0; r < 8; ++r) pos[r] = 0.f;

  const float* xlane =
      X + (size_t)(simBase + m16) * (NSTEP * IN_DIM) + ((lane & 16) ? 8 : 0);

  for (int t = 0; t < NSTEP - 1; ++t) {
    _Float16* hc = (t & 1) ? hbuf1 : hbuf0;   // current h (read)
    _Float16* hnx = (t & 1) ? hbuf0 : hbuf1;  // next h (write)

    // ---- gin A-fragment: K = [x(16), pos(8), zero-pad(8)] ----
    const float* xp = xlane + t * IN_DIM;
    v4f x0 = *(const v4f*)(xp);
    v4f x1 = *(const v4f*)(xp + 4);
    v8h lo;
    lo[0] = (_Float16)x0[0]; lo[1] = (_Float16)x0[1];
    lo[2] = (_Float16)x0[2]; lo[3] = (_Float16)x0[3];
    lo[4] = (_Float16)x1[0]; lo[5] = (_Float16)x1[1];
    lo[6] = (_Float16)x1[2]; lo[7] = (_Float16)x1[3];
    v8h hi = {};
    if (lane < 16) hi = *(const v8h*)(posrow + m16 * O_DIM);  // K=16..23 = pos
    v16h ginA = cat16(lo, hi);

    // ---- h A-fragments, reused across all 24 N-tiles ----
    v16h hA[KC];
#pragma unroll
    for (int c = 0; c < KC; ++c) hA[c] = load_hA(hc, lane, c);

    // ---- gates, 16 hidden units at a time (tiles u: r, u+8: z, u+16: n) ----
    for (int u = 0; u < 8; ++u) {
      v8f ar = {}, az = {}, ai = {}, ah = {};
      ar = wmma16(ginA, load_frag(sBih + (u + 0) * 512, lane), ar);
      az = wmma16(ginA, load_frag(sBih + (u + 8) * 512, lane), az);
      ai = wmma16(ginA, load_frag(sBih + (u + 16) * 512, lane), ai);
#pragma unroll
      for (int c = 0; c < KC; ++c) {
        ar = wmma16(hA[c], load_frag(sBhh + ((u + 0) * KC + c) * 512, lane), ar);
        az = wmma16(hA[c], load_frag(sBhh + ((u + 8) * KC + c) * 512, lane), az);
        ah = wmma16(hA[c], load_frag(sBhh + ((u + 16) * KC + c) * 512, lane), ah);
      }
      const int col = u * 16 + m16;
      const float bR = sBi[col] + sBh[col];
      const float bZ = sBi[H_DIM + col] + sBh[H_DIM + col];
      const float bI = sBi[2 * H_DIM + col];   // i_n bias
      const float bH = sBh[2 * H_DIM + col];   // h_n bias (gated by r)
#pragma unroll
      for (int r = 0; r < 8; ++r) {
        const int m = mb + r;
        float rr = fsig(ar[r] + bR);
        float zz = fsig(az[r] + bZ);
        float nn = ftanh_(ai[r] + bI + rr * (ah[r] + bH));
        float ho = (float)hc[m * HROW + col];
        float hv = (1.f - zz) * nn + zz * ho;  // torch GRUCell update
        hnx[m * HROW + col] = (_Float16)hv;
      }
    }

    // ---- output head: d = h_new @ W_out^T + b_out; clip vs pos; pos += d ----
    v8f ad = {};
#pragma unroll
    for (int c = 0; c < KC; ++c)
      ad = wmma16(load_hA(hnx, lane, c), load_frag(sBout + c * 512, lane), ad);

    if (m16 < O_DIM) {
      const float bo = sBo[m16];
#pragma unroll
      for (int r = 0; r < 8; ++r) {
        float d = ad[r] + bo;
        float p = pos[r];
        d = fminf(fmaxf(d, -1.f - p), 1.f - p);   // CAP = 1.0
        pos[r] = p + d;
        out[(size_t)(simBase + mb + r) * ((NSTEP - 1) * O_DIM) + t * O_DIM +
            m16] = d;
        posrow[(mb + r) * O_DIM + m16] = (_Float16)pos[r];  // stage for next gin
      }
    }
  }
}

extern "C" void kernel_launch(void* const* d_in, const int* in_sizes, int n_in,
                              void* d_out, int out_size, void* d_ws,
                              size_t ws_size, hipStream_t stream) {
  (void)n_in; (void)out_size; (void)d_ws; (void)ws_size;
  const float* X    = (const float*)d_in[0];
  const float* W_ih = (const float*)d_in[1];
  const float* W_hh = (const float*)d_in[2];
  const float* b_ih = (const float*)d_in[3];
  const float* b_hh = (const float*)d_in[4];
  const float* W_out = (const float*)d_in[5];
  const float* b_out = (const float*)d_in[6];
  float* out = (float*)d_out;

  const int nSim = in_sizes[0] / (NSTEP * IN_DIM);  // 16384
  const int blocks = nSim / SPB;                    // 128
  DeepHedging_GRU_wmma_kernel<<<dim3(blocks), dim3(256), 0, stream>>>(
      X, W_ih, W_hh, b_ih, b_hh, W_out, b_out, out);
}
